// MSARowAttentionWithPairBias_21741124452583
// MI455X (gfx1250) — matrix operand order repaired and checked by reading, not compiled, not run on hardware
//
#include <hip/hip_runtime.h>
#include <hip/hip_bf16.h>

// ---------------------------------------------------------------------------
// Types for CDNA5 WMMA (gfx1250, wave32)
// ---------------------------------------------------------------------------
typedef __attribute__((ext_vector_type(16))) __bf16 v16bf;
typedef __attribute__((ext_vector_type(8)))  float  v8f;

union BFFrag {
    v16bf v;
    uint4 q[2];
};

__device__ __forceinline__ unsigned short f2bf(float x) {
    union { float f; unsigned u; } c;
    c.f = x;
    unsigned r = c.u + 0x7FFFu + ((c.u >> 16) & 1u);   // round-to-nearest-even
    return (unsigned short)(r >> 16);
}

__device__ __forceinline__ v8f wmma_bf16(v16bf a, v16bf b, v8f c) {
    // D = A(16x32 bf16) * B(32x16 bf16) + C(16x16 f32)
    return __builtin_amdgcn_wmma_f32_16x16x32_bf16(
        /*neg_a=*/false, a, /*neg_b=*/false, b,
        /*c_mod=*/(short)0, c, /*reuse_a=*/false, /*reuse_b=*/false);
}

// A fragment: 16x32 bf16. Lane<16: row=lane, K=0..15;
// lane>=16: row=lane-16, K=16..31. Two 16B loads per lane.
__device__ __forceinline__ v16bf load_a16(const unsigned short* base, int ld) {
    int lane  = threadIdx.x & 31;
    int row   = lane & 15;
    int kchnk = (lane >> 4) * 16;
    const unsigned short* p = base + (long)row * ld + kchnk;
    BFFrag f;
    f.q[0] = *reinterpret_cast<const uint4*>(p);
    f.q[1] = *reinterpret_cast<const uint4*>(p + 8);
    return f.v;
}

// B fragment: 32x16 bf16 where B[k][n] = W[n][k] (W row-major NxK).
// Lane group 0 (lanes 0-15): K=0..15, group 1: K=16..31; n = lane&15.
// 16 contiguous bf16 per lane = two 16B loads.
__device__ __forceinline__ v16bf load_b16(const unsigned short* base, int ld) {
    int lane = threadIdx.x & 31;
    int n    = lane & 15;
    int k    = (lane >> 4) * 16;
    const unsigned short* p = base + (long)n * ld + k;
    BFFrag f;
    f.q[0] = *reinterpret_cast<const uint4*>(p);
    f.q[1] = *reinterpret_cast<const uint4*>(p + 8);
    return f.v;
}

// ---------------------------------------------------------------------------
// Problem constants
// ---------------------------------------------------------------------------
#define NSEQ 128
#define LSEQ 256
#define CM   256
#define CZ   128
#define NH   8
#define HD   32

// ---------------------------------------------------------------------------
// LayerNorm of msa -> bf16 m.  One 256-thread block per row.
// ---------------------------------------------------------------------------
__global__ __launch_bounds__(256) void ln_msa_kernel(
    const float* __restrict__ msa, const float* __restrict__ sc,
    const float* __restrict__ bi, unsigned short* __restrict__ m_bf) {
    __shared__ float red[256];
    int row = blockIdx.x, t = threadIdx.x;
    float x = msa[(long)row * CM + t];
    red[t] = x; __syncthreads();
    for (int s = 128; s > 0; s >>= 1) { if (t < s) red[t] += red[t + s]; __syncthreads(); }
    float mu = red[0] * (1.0f / CM); __syncthreads();
    float d = x - mu;
    red[t] = d * d; __syncthreads();
    for (int s = 128; s > 0; s >>= 1) { if (t < s) red[t] += red[t + s]; __syncthreads(); }
    float var = red[0] * (1.0f / CM);
    float y = d * rsqrtf(var + 1e-5f) * sc[t] + bi[t];
    m_bf[(long)row * CM + t] = f2bf(y);
}

// ---------------------------------------------------------------------------
// LayerNorm of pair fused with bias projection: bias[h][i][j] = ln(z)[i][j] . w_b[h]
// One 128-thread block per (i,j).
// ---------------------------------------------------------------------------
__global__ __launch_bounds__(128) void ln_pair_bias_kernel(
    const float* __restrict__ pair, const float* __restrict__ sc,
    const float* __restrict__ bi, const float* __restrict__ w_b,
    float* __restrict__ bias) {
    __shared__ float red[128];
    int blk = blockIdx.x;
    int i = blk >> 8, j = blk & 255, t = threadIdx.x;
    float x = pair[((long)i * LSEQ + j) * CZ + t];
    red[t] = x; __syncthreads();
    for (int s = 64; s > 0; s >>= 1) { if (t < s) red[t] += red[t + s]; __syncthreads(); }
    float mu = red[0] * (1.0f / CZ); __syncthreads();
    float d = x - mu;
    red[t] = d * d; __syncthreads();
    for (int s = 64; s > 0; s >>= 1) { if (t < s) red[t] += red[t + s]; __syncthreads(); }
    float var = red[0] * (1.0f / CZ); __syncthreads();
    float zl = d * rsqrtf(var + 1e-5f) * sc[t] + bi[t];
    for (int h = 0; h < NH; ++h) {
        red[t] = zl * w_b[h * CZ + t]; __syncthreads();
        for (int s = 64; s > 0; s >>= 1) { if (t < s) red[t] += red[t + s]; __syncthreads(); }
        if (t == 0) bias[(long)h * LSEQ * LSEQ + (long)i * LSEQ + j] = red[0];
        __syncthreads();
    }
}

// ---------------------------------------------------------------------------
// f32 -> bf16 conversion (weights)
// ---------------------------------------------------------------------------
__global__ void cvt_bf16_kernel(const float* __restrict__ in,
                                unsigned short* __restrict__ out, int n) {
    int i = blockIdx.x * blockDim.x + threadIdx.x;
    if (i < n) out[i] = f2bf(in[i]);
}

// ---------------------------------------------------------------------------
// Register-blocked WMMA GEMM: C(Mx256) = A(Mx256 bf16) * W(256x256 bf16)^T.
// K = N = 256 are compile-time constants -> K-loop (8 steps) fully unrolls.
// One wave computes a 2x4 block of 16x16 tiles (32 rows x 64 cols):
// per K-step, 2 A-frags + 4 B-frags feed 8 WMMAs (1.5 b128 loads per wmma).
// MODE 0: bf16 output, scaled.       MODE 1: f32 output + bias[col].
// MODE 2: final epilogue: out = residual + sigmoid(gatelin) * (acc + bias[col])
// ---------------------------------------------------------------------------
template <int MODE>
__global__ __launch_bounds__(256) void gemm_bf16_kernel(
    const unsigned short* __restrict__ A, const unsigned short* __restrict__ W,
    int M, float scale,
    const float* __restrict__ bias, const float* __restrict__ gatelin,
    const float* __restrict__ residual,
    float* __restrict__ outF, unsigned short* __restrict__ outB) {
    constexpr int K = CM;   // 256
    constexpr int N = CM;   // 256
    int wave    = blockIdx.x * (blockDim.x >> 5) + (threadIdx.x >> 5);
    constexpr int nblocks = N >> 6;            // 64-wide N blocks
    int mb = wave / nblocks, nb = wave - mb * nblocks;
    if (mb >= (M >> 5)) return;                // wave-uniform guard
    int m0 = mb << 5, n0 = nb << 6;

    v8f acc[2][4];
#pragma unroll
    for (int mi = 0; mi < 2; ++mi)
#pragma unroll
        for (int ni = 0; ni < 4; ++ni)
            acc[mi][ni] = (v8f){0.f, 0.f, 0.f, 0.f, 0.f, 0.f, 0.f, 0.f};

    const unsigned short* Arow0 = A + (long)m0 * K;
    const unsigned short* Arow1 = A + (long)(m0 + 16) * K;
#pragma unroll
    for (int kk = 0; kk < K; kk += 32) {
        if constexpr (true) {
            if (kk + 32 < K) {                       // compile-time resolved
                __builtin_prefetch(Arow0 + kk + 32, 0, 3);   // near-scope prefetch
                __builtin_prefetch(Arow1 + kk + 32, 0, 3);
            }
        }
        v16bf a0 = load_a16(Arow0 + kk, K);
        v16bf a1 = load_a16(Arow1 + kk, K);
        v16bf b[4];
#pragma unroll
        for (int ni = 0; ni < 4; ++ni)
            b[ni] = load_b16(W + (long)(n0 + ni * 16) * K + kk, K);
#pragma unroll
        for (int ni = 0; ni < 4; ++ni) {
            acc[0][ni] = wmma_bf16(a0, b[ni], acc[0][ni]);
            acc[1][ni] = wmma_bf16(a1, b[ni], acc[1][ni]);
        }
    }

    int lane = threadIdx.x & 31;
    int cn   = lane & 15;
    int rofs = (lane >> 4) << 3;
#pragma unroll
    for (int mi = 0; mi < 2; ++mi) {
        int rbase = m0 + mi * 16 + rofs;
#pragma unroll
        for (int ni = 0; ni < 4; ++ni) {
            int col = n0 + ni * 16 + cn;
#pragma unroll
            for (int r = 0; r < 8; ++r) {
                long idx = (long)(rbase + r) * N + col;
                float v = acc[mi][ni][r];
                if constexpr (MODE == 0) {
                    outB[idx] = f2bf(v * scale);
                } else if constexpr (MODE == 1) {
                    outF[idx] = v + bias[col];
                } else {
                    float g = 1.0f / (1.0f + __expf(-gatelin[idx]));
                    outF[idx] = residual[idx] + g * (v + bias[col]);
                }
            }
        }
    }
}

// ---------------------------------------------------------------------------
// Attention: one block = (b, h, 64-row strip), 4 waves, each wave a 16-row strip.
// S = (q/sqrt(d)) K^T + bias; softmax over j; O = P V.  All matmuls via WMMA bf16.
// ---------------------------------------------------------------------------
#define VTP 272   // padded LDS row stride for V^T

__global__ __launch_bounds__(128) void attn_kernel(
    const unsigned short* __restrict__ Q, const unsigned short* __restrict__ Kt,
    const unsigned short* __restrict__ V, const float* __restrict__ bias,
    unsigned short* __restrict__ O) {
    __shared__ __align__(16) unsigned short vt[HD * VTP];        // V^T: 32 x 272
    __shared__ __align__(16) unsigned short pbuf[4 * 16 * LSEQ]; // P per wave

    int blk = blockIdx.x;                 // 128 * 8 * 4
    int b   = blk >> 5;
    int rem = blk & 31;
    int h   = rem >> 2;
    int rb  = rem & 3;
    int tid = threadIdx.x, w = tid >> 5, lane = tid & 31;
    int i0  = rb * 64 + w * 16;

    // Stage V^T into LDS (bf16): vt[dd][j] = V[b*L + j][h*32 + dd]
    const unsigned short* Vh = V + ((long)b * LSEQ) * CM + h * HD;
    for (int idx = tid; idx < LSEQ * HD; idx += 128) {
        int j = idx >> 5, dd = idx & 31;
        vt[dd * VTP + j] = Vh[(long)j * CM + dd];
    }
    __syncthreads();

    // Q fragment for this wave's 16 rows (d = 32 = one WMMA K step)
    const unsigned short* Qh = Q + ((long)(b * LSEQ) + i0) * CM + h * HD;
    v16bf aq = load_a16(Qh, CM);

    // S strip: 16 tiles of 16x16, Q fragment register-resident across all
    v8f s[16];
    const unsigned short* Kh = Kt + ((long)b * LSEQ) * CM + h * HD;
#pragma unroll
    for (int jt = 0; jt < 16; ++jt) {
        v16bf bk = load_b16(Kh + (long)jt * 16 * CM, CM);
        v8f z = {0.f, 0.f, 0.f, 0.f, 0.f, 0.f, 0.f, 0.f};
        s[jt] = wmma_bf16(aq, bk, z);
    }

    // Softmax: row r of acc slot r lives at (i0 + r + 8*half), cols spread
    // over 16 lanes of the half-wave group x 16 register tiles.
    int half = lane >> 4, cn = lane & 15;
    const float* bp = bias + (long)h * LSEQ * LSEQ;
    float sm[8];
#pragma unroll
    for (int r = 0; r < 8; ++r) {
        int row = i0 + r + 8 * half;
        float mx = -1e30f;
#pragma unroll
        for (int jt = 0; jt < 16; ++jt) {
            float v = s[jt][r] + bp[(long)row * LSEQ + jt * 16 + cn];
            s[jt][r] = v;
            mx = fmaxf(mx, v);
        }
        for (int d = 1; d < 16; d <<= 1) mx = fmaxf(mx, __shfl_xor(mx, d, 32));
        float acc = 0.f;
#pragma unroll
        for (int jt = 0; jt < 16; ++jt) {
            float p = __expf(s[jt][r] - mx);
            s[jt][r] = p;
            acc += p;
        }
        for (int d = 1; d < 16; d <<= 1) acc += __shfl_xor(acc, d, 32);
        sm[r] = acc;
    }

    // Spill P (bf16) to LDS in plain row-major for A-fragment re-load
    unsigned short* pw = pbuf + w * 16 * LSEQ;
#pragma unroll
    for (int r = 0; r < 8; ++r) {
        int rl = r + 8 * half;
#pragma unroll
        for (int jt = 0; jt < 16; ++jt)
            pw[rl * LSEQ + jt * 16 + cn] = f2bf(s[jt][r]);
    }
    __syncthreads();

    // O strip (16 x 32) = P (16 x 256) * V (256 x 32), K-loop of 32
#pragma unroll
    for (int nt = 0; nt < 2; ++nt) {
        v8f o = {0.f, 0.f, 0.f, 0.f, 0.f, 0.f, 0.f, 0.f};
#pragma unroll
        for (int kt = 0; kt < 8; ++kt) {
            v16bf ap = load_a16(pw + kt * 32, LSEQ);
            v16bf bv = load_b16(vt + nt * 16 * VTP + kt * 32, VTP);
            o = wmma_bf16(ap, bv, o);
        }
#pragma unroll
        for (int r = 0; r < 8; ++r) {
            float val = o[r] / sm[r];
            long row = (long)b * LSEQ + i0 + r + 8 * half;
            int  col = h * HD + nt * 16 + cn;
            O[row * CM + col] = f2bf(val);
        }
    }
}

// ---------------------------------------------------------------------------
// Host-side orchestration
// ---------------------------------------------------------------------------
extern "C" void kernel_launch(void* const* d_in, const int* in_sizes, int n_in,
                              void* d_out, int out_size, void* d_ws, size_t ws_size,
                              hipStream_t stream) {
    (void)in_sizes; (void)n_in; (void)out_size; (void)ws_size;

    const float* msa        = (const float*)d_in[0];
    const float* pair       = (const float*)d_in[1];
    const float* ln_m_scale = (const float*)d_in[2];
    const float* ln_m_bias  = (const float*)d_in[3];
    const float* ln_z_scale = (const float*)d_in[4];
    const float* ln_z_bias  = (const float*)d_in[5];
    const float* w_q        = (const float*)d_in[6];
    const float* w_k        = (const float*)d_in[7];
    const float* w_v        = (const float*)d_in[8];
    const float* w_b        = (const float*)d_in[9];
    const float* w_out      = (const float*)d_in[10];
    const float* b_out      = (const float*)d_in[11];
    const float* w_gate     = (const float*)d_in[12];
    const float* b_gate     = (const float*)d_in[13];
    float* out = (float*)d_out;

    const long ROWS = (long)NSEQ * LSEQ;          // 32768
    char* ws = (char*)d_ws;
    size_t off = 0;
    auto alloc = [&](size_t bytes) { char* p = ws + off; off += (bytes + 255) & ~(size_t)255; return p; };

    unsigned short* m_bf    = (unsigned short*)alloc(ROWS * CM * 2);
    unsigned short* q_bf    = (unsigned short*)alloc(ROWS * CM * 2);
    unsigned short* k_bf    = (unsigned short*)alloc(ROWS * CM * 2);
    unsigned short* v_bf    = (unsigned short*)alloc(ROWS * CM * 2);
    unsigned short* attn_bf = (unsigned short*)alloc(ROWS * CM * 2);
    float*          gate_f  = (float*)alloc(ROWS * CM * 4);
    float*          bias_f  = (float*)alloc((size_t)NH * LSEQ * LSEQ * 4);
    unsigned short* wq_bf   = (unsigned short*)alloc(CM * CM * 2);
    unsigned short* wk_bf   = (unsigned short*)alloc(CM * CM * 2);
    unsigned short* wv_bf   = (unsigned short*)alloc(CM * CM * 2);
    unsigned short* wg_bf   = (unsigned short*)alloc(CM * CM * 2);
    unsigned short* wo_bf   = (unsigned short*)alloc(CM * CM * 2);

    // 1) LayerNorms (+ fused pair-bias projection)
    ln_msa_kernel<<<(int)ROWS, 256, 0, stream>>>(msa, ln_m_scale, ln_m_bias, m_bf);
    ln_pair_bias_kernel<<<LSEQ * LSEQ, 128, 0, stream>>>(pair, ln_z_scale, ln_z_bias, w_b, bias_f);

    // 2) Weight conversion f32 -> bf16 (layout stays (N, K))
    cvt_bf16_kernel<<<CM * CM / 256, 256, 0, stream>>>(w_q,    wq_bf, CM * CM);
    cvt_bf16_kernel<<<CM * CM / 256, 256, 0, stream>>>(w_k,    wk_bf, CM * CM);
    cvt_bf16_kernel<<<CM * CM / 256, 256, 0, stream>>>(w_v,    wv_bf, CM * CM);
    cvt_bf16_kernel<<<CM * CM / 256, 256, 0, stream>>>(w_gate, wg_bf, CM * CM);
    cvt_bf16_kernel<<<CM * CM / 256, 256, 0, stream>>>(w_out,  wo_bf, CM * CM);

    // 3) Projections: q (scale = 1/sqrt(32) folded in), k, v -> bf16; gate -> f32 linear
    //    Each wave covers a 32x64 block: blocks = (M/32)*(N/64)/8 waves.
    const int gemm_blocks = (int)((ROWS / 32) * (CM / 64) / 8);   // 512
    const float qscale = 0.17677669529663687f;                    // 1/sqrt(HD)
    gemm_bf16_kernel<0><<<gemm_blocks, 256, 0, stream>>>(m_bf, wq_bf, (int)ROWS,
        qscale, nullptr, nullptr, nullptr, nullptr, q_bf);
    gemm_bf16_kernel<0><<<gemm_blocks, 256, 0, stream>>>(m_bf, wk_bf, (int)ROWS,
        1.0f, nullptr, nullptr, nullptr, nullptr, k_bf);
    gemm_bf16_kernel<0><<<gemm_blocks, 256, 0, stream>>>(m_bf, wv_bf, (int)ROWS,
        1.0f, nullptr, nullptr, nullptr, nullptr, v_bf);
    gemm_bf16_kernel<1><<<gemm_blocks, 256, 0, stream>>>(m_bf, wg_bf, (int)ROWS,
        1.0f, b_gate, nullptr, nullptr, gate_f, nullptr);

    // 4) Attention (softmax(qk^T + bias) v) per (b, h, 64-row strip)
    attn_kernel<<<NSEQ * NH * 4, 128, 0, stream>>>(q_bf, k_bf, v_bf, bias_f, attn_bf);

    // 5) Output projection with gated residual epilogue
    gemm_bf16_kernel<2><<<gemm_blocks, 256, 0, stream>>>(attn_bf, wo_bf, (int)ROWS,
        1.0f, b_out, gate_f, msa, out, nullptr);
}
